// TreeLSTM_1975684956849
// MI455X (gfx1250) — matrix operand order repaired and checked by reading, not compiled, not run on hardware
//
#include <hip/hip_runtime.h>
#include <hip/hip_bf16.h>
#include <math.h>

// ---------------------------------------------------------------------------
// TreeLSTM on MI455X (gfx1250): level-batched GEMMs via v_wmma_f32_16x16x32_f16
// M=32 rows/block so every B fragment feeds two WMMAs (halves L2 weight traffic)
// ---------------------------------------------------------------------------

typedef __attribute__((ext_vector_type(16))) _Float16 v16h;
typedef __attribute__((ext_vector_type(8)))  _Float16 v8h;
typedef __attribute__((ext_vector_type(8)))  float    v8f;

#define TL_DEPTH 15
#define TL_N     65535      // 2^(DEPTH+1)-1
#define TL_H     256
#define TL_X     256
#define TL_3H    768
#define TL_C     5

#define STRH 264            // half-element LDS row stride for 256-wide arrays
#define STRW 776            // half-element LDS row stride for 768-wide iou staging

__device__ __forceinline__ float tl_sigmoid(float x) {
    return 1.0f / (1.0f + __expf(-x));
}

// 16x32 f16 A-fragment from LDS (row-major, stride STRH halves).
// ISA 16-bit A layout: lanes 0-15 rows 0-15 hold K=kb+0..7 (V0-3), K=kb+16..23
// (V4-7); lanes 16-31 same rows hold K=kb+8..15 / kb+24..31.
__device__ __forceinline__ v16h load_a_frag(const _Float16* s, int row, int kgrp, int kb) {
    const v8h* lo = (const v8h*)(s + row * STRH + kb + kgrp * 8);
    const v8h* hi = (const v8h*)(s + row * STRH + kb + 16 + kgrp * 8);
    v8h l = *lo;
    v8h h = *hi;
    v16h a;
#pragma unroll
    for (int i = 0; i < 8; ++i) { a[i] = l[i]; a[i + 8] = h[i]; }
    return a;
}

// B-fragment: weights pre-transposed to [n][k] f16; lane (col = lane&15,
// kgrp = lane>>4) loads K = kb+kgrp*8..+7 and kb+16+kgrp*8..+7.
__device__ __forceinline__ v16h load_b_frag(const _Float16* wT, int n0, int col, int kgrp, int kb) {
    const _Float16* p = wT + (size_t)(n0 + col) * 256 + kb + kgrp * 8;
    v8h l = *(const v8h*)p;
    v8h h = *(const v8h*)(p + 16);
    v16h b;
#pragma unroll
    for (int i = 0; i < 8; ++i) { b[i] = l[i]; b[i + 8] = h[i]; }
    return b;
}

__device__ __forceinline__ v8f wmma_f16(v16h a, v16h b, v8f c) {
    return __builtin_amdgcn_wmma_f32_16x16x32_f16(false, a, false, b, (short)0, c, false, false);
}

// ---------------------------------------------------------------------------
// Weight conversion: fp32 row-major [K][N] -> f16 n-major [N][K]
// ---------------------------------------------------------------------------
__global__ void tl_convert_weights(const float* __restrict__ W_iou,
                                   const float* __restrict__ U_iou,
                                   const float* __restrict__ W_f,
                                   const float* __restrict__ U_f,
                                   _Float16* __restrict__ WiouT,
                                   _Float16* __restrict__ UiouT,
                                   _Float16* __restrict__ WfT,
                                   _Float16* __restrict__ UfT) {
    int i = blockIdx.x * blockDim.x + threadIdx.x;
    if (i < TL_3H * 256) {
        int n = i / 256, k = i % 256;
        WiouT[i] = (_Float16)W_iou[k * TL_3H + n];
        UiouT[i] = (_Float16)U_iou[k * TL_3H + n];
    } else {
        int j = i - TL_3H * 256;
        if (j < 256 * 256) {
            int n = j / 256, k = j % 256;
            WfT[j] = (_Float16)W_f[k * 256 + n];
            UfT[j] = (_Float16)U_f[k * 256 + n];
        }
    }
}

// ---------------------------------------------------------------------------
// One tree level. Each block = 32 nodes (two 16-row WMMA tiles), 512 threads
// (16 waves). Tile jobs (each job covers both row-tiles, sharing B loads):
//   jobs  0..47 : iou = x@W_iou + (hl+hr)@U_iou   (48 n-tiles of 16 = 768)
//   jobs 48..63 : fx  = x@W_f                      (16 n-tiles = 256)
//   jobs 64..79 : ul  = hl@U_f
//   jobs 80..95 : ur  = hr@U_f
// K = 256 swept in 8 WMMA(16x16x32) per operand pair per row-tile.
// (hl+hr) A-fragments are formed in VGPRs: hl_frag + hr_frag.
// ---------------------------------------------------------------------------
template <bool LEAF>
__global__ void __launch_bounds__(512)
tl_level_kernel(const int* __restrict__ tokens,
                const float* __restrict__ emb,
                const _Float16* __restrict__ WiouT,
                const _Float16* __restrict__ UiouT,
                const _Float16* __restrict__ WfT,
                const _Float16* __restrict__ UfT,
                const float* __restrict__ b_iou,
                const float* __restrict__ b_f,
                float* __restrict__ h,
                float* __restrict__ c,
                int level_base, int n_l) {
    __shared__ _Float16 xs[32 * STRH];
    __shared__ _Float16 hls[LEAF ? 1 : 32 * STRH];
    __shared__ _Float16 hrs[LEAF ? 1 : 32 * STRH];
    __shared__ _Float16 iou_s[32 * STRW];            // f16 staging of iou pre-act
    __shared__ _Float16 fx_s[LEAF ? 1 : 32 * STRH];
    __shared__ _Float16 ul_s[LEAF ? 1 : 32 * STRH];
    __shared__ _Float16 ur_s[LEAF ? 1 : 32 * STRH];

    const int tid  = threadIdx.x;
    const int row0 = blockIdx.x * 32;

    // ---- stage x = emb[tokens[idx]] as f16 (rows clamped for partial tiles)
    for (int e = tid; e < 32 * 256; e += 512) {
        int r = e >> 8, k = e & 255;
        int node = row0 + r;
        if (node > n_l - 1) node = n_l - 1;
        int tok = tokens[level_base + node];
        xs[r * STRH + k] = (_Float16)emb[(size_t)tok * TL_X + k];
    }

    // ---- stage children h (fp32 -> f16)
    if constexpr (!LEAF) {
        for (int e = tid; e < 32 * 256; e += 512) {
            int r = e >> 8, k = e & 255;
            int node = row0 + r;
            if (node > n_l - 1) node = n_l - 1;
            int g  = level_base + node;
            int li = 2 * g + 1, ri = 2 * g + 2;
            hls[r * STRH + k] = (_Float16)h[(size_t)li * TL_H + k];
            hrs[r * STRH + k] = (_Float16)h[(size_t)ri * TL_H + k];
        }
    }
    __syncthreads();

    // ---- WMMA tile sweep
    const int wave = tid >> 5;
    const int lane = tid & 31;
    const int col  = lane & 15;   // B column / A row within the fragment
    const int kgrp = lane >> 4;   // 0 or 1
    const int nT   = LEAF ? 48 : 96;

    for (int t = wave; t < nT; t += 16) {
        v8f acc0 = {0.f, 0.f, 0.f, 0.f, 0.f, 0.f, 0.f, 0.f};
        v8f acc1 = acc0;
        if (t < 48) {
            int n0 = t * 16;
#pragma unroll
            for (int kb = 0; kb < 256; kb += 32) {
                v16h b  = load_b_frag(WiouT, n0, col, kgrp, kb);
                acc0 = wmma_f16(load_a_frag(xs, col,      kgrp, kb), b, acc0);
                acc1 = wmma_f16(load_a_frag(xs, col + 16, kgrp, kb), b, acc1);
            }
            if constexpr (!LEAF) {
#pragma unroll
                for (int kb = 0; kb < 256; kb += 32) {
                    v16h b  = load_b_frag(UiouT, n0, col, kgrp, kb);
                    v16h a0 = load_a_frag(hls, col, kgrp, kb)
                            + load_a_frag(hrs, col, kgrp, kb);       // child-sum in VGPRs
                    v16h a1 = load_a_frag(hls, col + 16, kgrp, kb)
                            + load_a_frag(hrs, col + 16, kgrp, kb);
                    acc0 = wmma_f16(a0, b, acc0);
                    acc1 = wmma_f16(a1, b, acc1);
                }
            }
            // C/D layout: VGPR r -> row (r + 8*kgrp), column col
#pragma unroll
            for (int r = 0; r < 8; ++r) {
                iou_s[(kgrp * 8 + r) * STRW + n0 + col]        = (_Float16)acc0[r];
                iou_s[(16 + kgrp * 8 + r) * STRW + n0 + col]   = (_Float16)acc1[r];
            }
        } else if constexpr (!LEAF) {
            int grp = (t - 48) >> 4;             // 0: fx, 1: ul, 2: ur
            int n0  = ((t - 48) & 15) * 16;
            const _Float16* A  = (grp == 0) ? xs  : (grp == 1) ? hls : hrs;
            const _Float16* Bw = (grp == 0) ? WfT : UfT;
            _Float16* dst      = (grp == 0) ? fx_s : (grp == 1) ? ul_s : ur_s;
#pragma unroll
            for (int kb = 0; kb < 256; kb += 32) {
                v16h b = load_b_frag(Bw, n0, col, kgrp, kb);
                acc0 = wmma_f16(load_a_frag(A, col,      kgrp, kb), b, acc0);
                acc1 = wmma_f16(load_a_frag(A, col + 16, kgrp, kb), b, acc1);
            }
#pragma unroll
            for (int r = 0; r < 8; ++r) {
                dst[(kgrp * 8 + r) * STRH + n0 + col]      = (_Float16)acc0[r];
                dst[(16 + kgrp * 8 + r) * STRH + n0 + col] = (_Float16)acc1[r];
            }
        }
    }
    __syncthreads();

    // ---- elementwise gates, write h/c
    for (int e = tid; e < 32 * 256; e += 512) {
        int r = e >> 8, j = e & 255;
        int node = row0 + r;
        if (node >= n_l) continue;
        int g = level_base + node;
        float iv = (float)iou_s[r * STRW + j]       + b_iou[j];
        float ov = (float)iou_s[r * STRW + 256 + j] + b_iou[256 + j];
        float uv = (float)iou_s[r * STRW + 512 + j] + b_iou[512 + j];
        float cn = tl_sigmoid(iv) * tanhf(uv);
        if constexpr (!LEAF) {
            int li = 2 * g + 1, ri = 2 * g + 2;
            float fx = (float)fx_s[r * STRH + j] + b_f[j];
            float fl = tl_sigmoid(fx + (float)ul_s[r * STRH + j]);
            float fr = tl_sigmoid(fx + (float)ur_s[r * STRH + j]);
            cn += fl * c[(size_t)li * TL_H + j] + fr * c[(size_t)ri * TL_H + j];
        }
        float hn = tl_sigmoid(ov) * tanhf(cn);
        c[(size_t)g * TL_H + j] = cn;
        h[(size_t)g * TL_H + j] = hn;
    }
}

// ---------------------------------------------------------------------------
// Readout: mean NLL over all nodes (tiny 256x5 matvec per node)
// ---------------------------------------------------------------------------
__global__ void tl_zero_out(float* out) { out[0] = 0.0f; }

__global__ void tl_readout(const float* __restrict__ h,
                           const int* __restrict__ labels,
                           const float* __restrict__ W_out,
                           const float* __restrict__ b_out,
                           float* __restrict__ out) {
    int n   = blockIdx.x * blockDim.x + threadIdx.x;
    int tid = threadIdx.x;
    float local = 0.0f;
    if (n < TL_N) {
        float acc[TL_C];
#pragma unroll
        for (int cc = 0; cc < TL_C; ++cc) acc[cc] = b_out[cc];
        for (int j = 0; j < TL_H; ++j) {
            float hv = h[(size_t)n * TL_H + j];
#pragma unroll
            for (int cc = 0; cc < TL_C; ++cc) acc[cc] += hv * W_out[j * TL_C + cc];
        }
        float m = acc[0];
#pragma unroll
        for (int cc = 1; cc < TL_C; ++cc) m = fmaxf(m, acc[cc]);
        float s = 0.0f;
#pragma unroll
        for (int cc = 0; cc < TL_C; ++cc) s += __expf(acc[cc] - m);
        int lab = labels[n];
        local = -(acc[lab] - m - __logf(s));
    }
    __shared__ float red[256];
    red[tid] = local;
    __syncthreads();
#pragma unroll
    for (int stp = 128; stp > 0; stp >>= 1) {
        if (tid < stp) red[tid] += red[tid + stp];
        __syncthreads();
    }
    if (tid == 0) atomicAdd(out, red[0] * (1.0f / (float)TL_N));
}

// ---------------------------------------------------------------------------
extern "C" void kernel_launch(void* const* d_in, const int* in_sizes, int n_in,
                              void* d_out, int out_size, void* d_ws, size_t ws_size,
                              hipStream_t stream) {
    const int*   tokens = (const int*)d_in[0];
    const int*   labels = (const int*)d_in[1];
    const float* emb    = (const float*)d_in[2];
    const float* W_iou  = (const float*)d_in[3];
    const float* U_iou  = (const float*)d_in[4];
    const float* b_iou  = (const float*)d_in[5];
    const float* W_f    = (const float*)d_in[6];
    const float* U_f    = (const float*)d_in[7];
    const float* b_f    = (const float*)d_in[8];
    const float* W_out  = (const float*)d_in[9];
    const float* b_out  = (const float*)d_in[10];

    // Workspace layout
    float* h = (float*)d_ws;
    float* c = h + (size_t)TL_N * TL_H;
    _Float16* WiouT = (_Float16*)(c + (size_t)TL_N * TL_H);
    _Float16* UiouT = WiouT + (size_t)TL_3H * 256;
    _Float16* WfT   = UiouT + (size_t)TL_3H * 256;
    _Float16* UfT   = WfT   + (size_t)256 * 256;

    // 1) convert + transpose weights to f16 n-major
    {
        int total  = TL_3H * 256 + 256 * 256;  // 262144
        int blocks = (total + 255) / 256;
        tl_convert_weights<<<blocks, 256, 0, stream>>>(W_iou, U_iou, W_f, U_f,
                                                       WiouT, UiouT, WfT, UfT);
    }

    // 2) levels bottom-up
    for (int l = TL_DEPTH; l >= 0; --l) {
        int n_l    = 1 << l;
        int base   = n_l - 1;
        int blocks = (n_l + 31) / 32;
        if (l == TL_DEPTH) {
            tl_level_kernel<true><<<blocks, 512, 0, stream>>>(
                tokens, emb, WiouT, UiouT, WfT, UfT, b_iou, b_f, h, c, base, n_l);
        } else {
            tl_level_kernel<false><<<blocks, 512, 0, stream>>>(
                tokens, emb, WiouT, UiouT, WfT, UfT, b_iou, b_f, h, c, base, n_l);
        }
    }

    // 3) readout loss
    tl_zero_out<<<1, 1, 0, stream>>>((float*)d_out);
    tl_readout<<<(TL_N + 255) / 256, 256, 0, stream>>>(h, labels, W_out, b_out,
                                                       (float*)d_out);
}
